// UnitGameNet_979252543573
// MI455X (gfx1250) — matrix operand order repaired and checked by reading, not compiled, not run on hardware
//
#include <hip/hip_runtime.h>
#include <hip/hip_bf16.h>

typedef __attribute__((ext_vector_type(16))) _Float16 v16h;
typedef __attribute__((ext_vector_type(4)))  _Float16 v4h;
typedef __attribute__((ext_vector_type(8)))  float    v8f;

#define BB   32
#define NN   8192
#define FIN  5
#define EE   131072
#define CC   32
#define PC   4
#define BN   (BB * NN)          // 262144 nodes total
#define NLAY 5

// ---------------- workspace layout (bytes) ----------------
#define OFF_H      0u                       // f16 [BN*32]      16,777,216
#define OFF_S      16777216u                // f32 [BN*32]      33,554,432
#define OFF_AGG    50331648u                // f32 [BN*32]      33,554,432
// t (f16 [BN*128] = 67,108,864 B) aliases OFF_S..OFF_AGG end (GCN scratch dead by then)
#define OFF_T      OFF_S
#define OFF_LOG    83886080u                // f32 [BN*4]        4,194,304
#define OFF_DINV   88080384u                // f32 [N]
#define OFF_DEG    88113152u                // f32 [N]
#define OFF_WTS    88145920u                // f16 weights (~22.5 KB)
// wts layout (halves): gcnWt l*1024 (l=0..4), pW1t @5120 (4096), pW2t @9216 (2048)

union AFrag { v16h v; uint4 q[2]; unsigned u[8]; };

// ---------------- prep kernels ----------------
__global__ void k_zero(float* p, int n) {
  int i = blockIdx.x * 256 + threadIdx.x;
  if (i < n) p[i] = 0.f;
}

__global__ void k_deg(const int* __restrict__ col, float* __restrict__ deg) {
  int e = blockIdx.x * 256 + threadIdx.x;
  if (e < EE) atomicAdd(&deg[col[e]], 1.0f);
}

__global__ void k_dinv(const float* __restrict__ deg, float* __restrict__ dinv) {
  int n = blockIdx.x * 256 + threadIdx.x;
  if (n < NN) dinv[n] = rsqrtf(deg[n] + 1.0f);
}

// Transpose weights to f16 so the WMMA B-operand (K contiguous per column) loads as b128s.
__global__ void k_prep(const float* W0, const float* W1, const float* W2,
                       const float* W3, const float* W4,
                       const float* pW1, const float* pW2, _Float16* wts) {
  int tid = threadIdx.x;
  const float* Ws[5] = {W0, W1, W2, W3, W4};
  for (int l = 0; l < 5; ++l) {
    _Float16* Wt = wts + l * 1024;            // Wt[o*32 + i] = W[i][o], K-padded
    int kin = (l == 0) ? FIN : CC;
    for (int idx = tid; idx < 1024; idx += blockDim.x) {
      int o = idx >> 5, i = idx & 31;
      Wt[o * 32 + i] = (i < kin) ? (_Float16)Ws[l][i * 32 + o] : (_Float16)0.f;
    }
  }
  _Float16* p1 = wts + 5120;                  // p1[o*32 + i] = pW1[i][o]
  for (int idx = tid; idx < 4096; idx += blockDim.x) {
    int o = idx >> 5, i = idx & 31;
    p1[o * 32 + i] = (_Float16)pW1[i * 128 + o];
  }
  _Float16* p2 = wts + 9216;                  // p2[n*128 + k] = pW2[k][n], N-padded 4->16
  for (int idx = tid; idx < 2048; idx += blockDim.x) {
    int n = idx >> 7, k = idx & 127;
    p2[n * 128 + k] = (n < PC) ? (_Float16)pW2[k * PC + n] : (_Float16)0.f;
  }
}

// Pack x [BN,5] f32 -> h0 [BN,32] f16 (zero-pad K)
__global__ void k_h0(const float* __restrict__ x, _Float16* __restrict__ h) {
  int i = blockIdx.x * 256 + threadIdx.x;    // BN*32 threads
  int c = i & 31, node = i >> 5;
  h[i] = (c < FIN) ? (_Float16)x[node * FIN + c] : (_Float16)0.f;
}

// ---------------- WMMA A-fragment: 16x32 f16 tile at rows [row0,row0+16) ----------------
// ISA 7.12.2 layout: per lane (m=lane&15, kh=lane>>4) the 8 dwords are row bytes
// [kh*16, kh*16+16) and [32+kh*16, 32+kh*16+16)  ->  two global_load_b128.
__device__ __forceinline__ v16h load_a(const _Float16* __restrict__ base, int row0,
                                       int row_stride_h, int lane) {
  int m = lane & 15, kh = lane >> 4;
  AFrag A;
  const uint4* p = (const uint4*)(base + (size_t)(row0 + m) * row_stride_h);
  A.q[0] = p[kh];
  A.q[1] = p[2 + kh];
  return A.v;
}

// B-fragment 32x16 f16 from transposed weights Wt[n][k]: K [8kg,8kg+8) and [16+8kg,24+8kg) -> two b128.
__device__ __forceinline__ v16h load_b(const _Float16* __restrict__ Wt, int ncol0,
                                       int col_stride_h, int koff, int lane) {
  int n = lane & 15, kg = lane >> 4;          // lanes 0-15: K 0..7,16..23; 16-31: K 8..15,24..31
  AFrag B;
  const uint4* p = (const uint4*)(Wt + (size_t)(ncol0 + n) * col_stride_h + koff);
  B.q[0] = p[kg];
  B.q[1] = p[2 + kg];
  return B.v;
}

// ---------------- GCN matmul: s = (h @ W) * dinv ; agg = s ----------------
__global__ void k_mm(const _Float16* __restrict__ h, const _Float16* __restrict__ Wt,
                     const float* __restrict__ dinv,
                     float* __restrict__ s, float* __restrict__ agg) {
  int lane = threadIdx.x & 31, wave = threadIdx.x >> 5;
  int row0 = blockIdx.x * 128 + wave * 16;
  v16h a = load_a(h, row0, 32, lane);
  int nloc = lane & 15, mb = (lane >> 4) * 8;
#pragma unroll
  for (int j = 0; j < 2; ++j) {
    v16h b = load_b(Wt, j * 16, 32, 0, lane);
    v8f c = {};
    c = __builtin_amdgcn_wmma_f32_16x16x32_f16(false, a, false, b, (short)0, c, false, false);
#pragma unroll
    for (int r = 0; r < 8; ++r) {
      int row = row0 + mb + r;
      float v = c[r] * dinv[row & (NN - 1)];
      int off = row * 32 + j * 16 + nloc;
      s[off] = v;
      agg[off] = v;                            // A_hat self-loop term
    }
  }
}

// ---------------- edge scatter: agg[col] += s[row]  (4 channels / thread) ----------------
__global__ void k_edge(const float* __restrict__ s, float* __restrict__ agg,
                       const int* __restrict__ row, const int* __restrict__ col) {
  unsigned idx = blockIdx.x * 256u + threadIdx.x;   // BB*EE*8 threads
  unsigned e = idx >> 3;
  int ch = (idx & 7) * 4;
  int b = e >> 17;                                  // EE = 2^17
  int eg = e & (EE - 1);
  int r = row[eg] + b * NN;
  int c = col[eg] + b * NN;
  const float4 v = *(const float4*)(s + (size_t)r * 32 + ch);
  float* d = agg + (size_t)c * 32 + ch;
  atomicAdd(d + 0, v.x); atomicAdd(d + 1, v.y);
  atomicAdd(d + 2, v.z); atomicAdd(d + 3, v.w);
}

// ---------------- h = relu(agg * dinv + bias), back to f16 (4 chans / thread) ----------------
__global__ void k_relu(const float* __restrict__ agg, const float* __restrict__ dinv,
                       const float* __restrict__ bias, _Float16* __restrict__ h) {
  int i4 = (blockIdx.x * 256 + threadIdx.x) * 4;   // BN*32/4 threads
  int c0 = i4 & 31, node = i4 >> 5;
  float di = dinv[node & (NN - 1)];
  const float4 a = *(const float4*)(agg + i4);
  const float4 bi = *(const float4*)(bias + c0);
  v4h o;
  o.x = (_Float16)fmaxf(a.x * di + bi.x, 0.f);
  o.y = (_Float16)fmaxf(a.y * di + bi.y, 0.f);
  o.z = (_Float16)fmaxf(a.z * di + bi.z, 0.f);
  o.w = (_Float16)fmaxf(a.w * di + bi.w, 0.f);
  *(v4h*)(h + i4) = o;
}

// ---------------- policy layer 1: t = relu(h @ pW1 + pb1)  [BN,128] f16 ----------------
__global__ void k_pol1(const _Float16* __restrict__ h, const _Float16* __restrict__ Wt,
                       const float* __restrict__ bias, _Float16* __restrict__ t) {
  int lane = threadIdx.x & 31, wave = threadIdx.x >> 5;
  int row0 = blockIdx.x * 128 + wave * 16;
  v16h a = load_a(h, row0, 32, lane);
  int nloc = lane & 15, mb = (lane >> 4) * 8;
#pragma unroll
  for (int j = 0; j < 8; ++j) {
    v16h b = load_b(Wt, j * 16, 32, 0, lane);
    v8f c = {};
    c = __builtin_amdgcn_wmma_f32_16x16x32_f16(false, a, false, b, (short)0, c, false, false);
    int colg = j * 16 + nloc;
    float bi = bias[colg];
#pragma unroll
    for (int r = 0; r < 8; ++r) {
      int row = row0 + mb + r;
      t[(size_t)row * 128 + colg] = (_Float16)fmaxf(c[r] + bi, 0.f);
    }
  }
}

// ---------------- policy layer 2: logits = t @ pW2 + pb2  [BN,4] f32 ----------------
__global__ void k_pol2(const _Float16* __restrict__ t, const _Float16* __restrict__ Wt,
                       const float* __restrict__ bias, float* __restrict__ logits) {
  int lane = threadIdx.x & 31, wave = threadIdx.x >> 5;
  int row0 = blockIdx.x * 128 + wave * 16;
  int nloc = lane & 15, mb = (lane >> 4) * 8;
  int m = lane & 15, kh = lane >> 4;
  const uint4* tp = (const uint4*)(t + (size_t)(row0 + m) * 128);  // 16 uint4 per row
  v8f c = {};
#pragma unroll
  for (int kk = 0; kk < 4; ++kk) {                  // K = 128 = 4 x 32
    AFrag A;
    A.q[0] = tp[kk * 4 + kh];
    A.q[1] = tp[kk * 4 + 2 + kh];
    v16h b = load_b(Wt, 0, 128, kk * 32, lane);
    c = __builtin_amdgcn_wmma_f32_16x16x32_f16(false, A.v, false, b, (short)0, c, false, false);
  }
  if (nloc < PC) {
    float bi = bias[nloc];
#pragma unroll
    for (int r = 0; r < 8; ++r) {
      int row = row0 + mb + r;
      logits[(size_t)row * PC + nloc] = c[r] + bi;
    }
  }
}

// ---------------- softmax over N*PC = 32768 per graph ----------------
__global__ void k_softmax(const float* __restrict__ lg, float* __restrict__ out) {
  __shared__ float red[256];
  int b = blockIdx.x, tid = threadIdx.x;
  const float* p = lg + (size_t)b * 32768;
  float* o = out + (size_t)b * 32768;
  float m = -1e30f;
  for (int j = tid; j < 32768; j += 256) m = fmaxf(m, p[j]);
  red[tid] = m; __syncthreads();
  for (int s2 = 128; s2 > 0; s2 >>= 1) { if (tid < s2) red[tid] = fmaxf(red[tid], red[tid + s2]); __syncthreads(); }
  float M = red[0]; __syncthreads();
  float sum = 0.f;
  for (int j = tid; j < 32768; j += 256) sum += expf(p[j] - M);
  red[tid] = sum; __syncthreads();
  for (int s2 = 128; s2 > 0; s2 >>= 1) { if (tid < s2) red[tid] += red[tid + s2]; __syncthreads(); }
  float inv = 1.0f / red[0]; __syncthreads();
  for (int j = tid; j < 32768; j += 256) o[j] = expf(p[j] - M) * inv;
}

// ---------------- pooled mean + value MLP + tanh, one block per graph ----------------
__global__ void k_value(const _Float16* __restrict__ h,
                        const float* __restrict__ vW1, const float* __restrict__ vb1,
                        const float* __restrict__ vW2, const float* __restrict__ vb2,
                        const float* __restrict__ vW3, const float* __restrict__ vb3,
                        float* __restrict__ out) {
  __shared__ float red[256];
  __shared__ float pooled[32];
  __shared__ float v1[256];
  __shared__ float v2[64];
  int b = blockIdx.x, tid = threadIdx.x;
  float acc[32];
#pragma unroll
  for (int c = 0; c < 32; ++c) acc[c] = 0.f;
  const _Float16* hb = h + (size_t)b * NN * 32;
  for (int n = tid; n < NN; n += 256) {
    const _Float16* hr = hb + (size_t)n * 32;
#pragma unroll
    for (int c = 0; c < 32; ++c) acc[c] += (float)hr[c];
  }
  for (int c = 0; c < 32; ++c) {
    red[tid] = acc[c]; __syncthreads();
    for (int s2 = 128; s2 > 0; s2 >>= 1) { if (tid < s2) red[tid] += red[tid + s2]; __syncthreads(); }
    if (tid == 0) pooled[c] = red[0] * (1.0f / (float)NN);
    __syncthreads();
  }
  // v1 = relu(pooled @ vW1 + vb1)   [256]
  {
    float s = vb1[tid];
    for (int c = 0; c < 32; ++c) s += pooled[c] * vW1[c * 256 + tid];
    v1[tid] = fmaxf(s, 0.f);
  }
  __syncthreads();
  // v2 = relu(v1 @ vW2 + vb2)       [64]
  if (tid < 64) {
    float s = vb2[tid];
    for (int k = 0; k < 256; ++k) s += v1[k] * vW2[k * 64 + tid];
    v2[tid] = fmaxf(s, 0.f);
  }
  __syncthreads();
  if (tid == 0) {
    float s = vb3[0];
    for (int k = 0; k < 64; ++k) s += v2[k] * vW3[k];
    out[b] = tanhf(s);
  }
}

extern "C" void kernel_launch(void* const* d_in, const int* in_sizes, int n_in,
                              void* d_out, int out_size, void* d_ws, size_t ws_size,
                              hipStream_t stream) {
  // inputs: dict order, tuples flattened
  const float* x   = (const float*)d_in[0];
  const int*   ei  = (const int*)d_in[1];          // [2,E] -> row=ei[0..E), col=ei[E..2E)
  const float* gW[5] = {(const float*)d_in[2], (const float*)d_in[3], (const float*)d_in[4],
                        (const float*)d_in[5], (const float*)d_in[6]};
  const float* gB[5] = {(const float*)d_in[7], (const float*)d_in[8], (const float*)d_in[9],
                        (const float*)d_in[10], (const float*)d_in[11]};
  const float* pW1 = (const float*)d_in[12];
  const float* pb1 = (const float*)d_in[13];
  const float* pW2 = (const float*)d_in[14];
  const float* pb2 = (const float*)d_in[15];
  const float* vW1 = (const float*)d_in[16];
  const float* vb1 = (const float*)d_in[17];
  const float* vW2 = (const float*)d_in[18];
  const float* vb2 = (const float*)d_in[19];
  const float* vW3 = (const float*)d_in[20];
  const float* vb3 = (const float*)d_in[21];

  char* ws = (char*)d_ws;
  _Float16* h      = (_Float16*)(ws + OFF_H);
  float*    s      = (float*)(ws + OFF_S);
  float*    agg    = (float*)(ws + OFF_AGG);
  _Float16* t      = (_Float16*)(ws + OFF_T);
  float*    logits = (float*)(ws + OFF_LOG);
  float*    dinv   = (float*)(ws + OFF_DINV);
  float*    deg    = (float*)(ws + OFF_DEG);
  _Float16* wts    = (_Float16*)(ws + OFF_WTS);
  float*    out    = (float*)d_out;

  const int* row = ei;
  const int* col = ei + EE;

  k_zero<<<(NN + 255) / 256, 256, 0, stream>>>(deg, NN);
  k_deg<<<EE / 256, 256, 0, stream>>>(col, deg);
  k_dinv<<<NN / 256, 256, 0, stream>>>(deg, dinv);
  k_prep<<<1, 256, 0, stream>>>(gW[0], gW[1], gW[2], gW[3], gW[4], pW1, pW2, wts);
  k_h0<<<(BN * 32) / 256, 256, 0, stream>>>(x, h);

  for (int l = 0; l < NLAY; ++l) {
    k_mm<<<BN / 128, 256, 0, stream>>>(h, wts + l * 1024, dinv, s, agg);
    k_edge<<<(BB * EE * 8) / 256, 256, 0, stream>>>(s, agg, row, col);
    k_relu<<<(BN * 32 / 4) / 256, 256, 0, stream>>>(agg, dinv, gB[l], h);
  }

  k_pol1<<<BN / 128, 256, 0, stream>>>(h, wts + 5120, pb1, t);
  k_pol2<<<BN / 128, 256, 0, stream>>>(t, wts + 9216, pb2, logits);
  k_softmax<<<BB, 256, 0, stream>>>(logits, out);
  k_value<<<BB, 256, 0, stream>>>(h, vW1, vb1, vW2, vb2, vW3, vb3, out + (size_t)BB * NN * PC);
}
